// SchemaItemClassifier_42047729827911
// MI455X (gfx1250) — matrix operand order repaired and checked by reading, not compiled, not run on hardware
//
#include <hip/hip_runtime.h>
#include <hip/hip_bf16.h>
#include <math.h>

// Problem constants (match reference)
#define BB  4
#define SS  4096
#define HH  1024
#define TT  32
#define CCC 8
#define LTT 5
#define LCC 7
#define HCC 512        // HC = H/2
#define NHH 8
#define DHH 128

typedef __attribute__((ext_vector_type(16))) _Float16 v16h;
typedef __attribute__((ext_vector_type(8)))  float    v8f;

// ---------------------------------------------------------------------------
// WMMA fragment loaders (wave32, V_WMMA_F32_16X16X32_F16 layouts per ISA 7.12.2)
// ---------------------------------------------------------------------------
// A: 16x32 tile of row-major f16 X.
//   element e of lane L: m = m0 + (L&15), k = k0 + ((e>>3)<<4) + ((L>>4)<<3) + (e&7)
//   -> per lane: two contiguous 8-halfword (16B) chunks at kb and kb+16.
__device__ __forceinline__ v16h load_A16(const _Float16* __restrict__ X, int ldx,
                                         int m0, int k0, int lane) {
  const _Float16* row = X + (size_t)(m0 + (lane & 15)) * ldx + (k0 + ((lane >> 4) << 3));
  v16h a;
#pragma unroll
  for (int e = 0; e < 8; ++e) a[e] = row[e];
#pragma unroll
  for (int e = 0; e < 8; ++e) a[8 + e] = row[16 + e];
  return a;
}

// B: 32x16 tile B(k,n) = W[n][k], W row-major (N x K) f16:
//   lane L: n = n0 + (L&15), k = k0 + ((e>>3)<<4) + ((L>>4)<<3) + (e&7) -> two 8-f16 (16B) chunks
__device__ __forceinline__ v16h load_B16(const _Float16* __restrict__ W, int ldw,
                                         int n0, int k0, int lane) {
  const _Float16* p = W + (size_t)(n0 + (lane & 15)) * ldw + (k0 + ((lane >> 4) << 3));
  v16h b;
#pragma unroll
  for (int e = 0; e < 8; ++e) b[e] = p[e];
#pragma unroll
  for (int e = 0; e < 8; ++e) b[8 + e] = p[16 + e];
  return b;
}

// C/D layout: VGPR v -> m = m0 + v + 8*(lane>>4); n = n0 + (lane&15)
__device__ __forceinline__ void store_tile(v8f acc, float* __restrict__ Y,
                                           _Float16* __restrict__ Y16,
                                           const float* __restrict__ bias,
                                           const float* __restrict__ residual,
                                           int N, int m0, int n0, int lane, int act) {
  const int n = n0 + (lane & 15);
  const int mb = m0 + ((lane >> 4) << 3);
  const float bv = bias ? bias[n] : 0.0f;
#pragma unroll
  for (int v = 0; v < 8; ++v) {
    const int m = mb + v;
    float y = acc[v] + bv;
    if (act == 1) y = (y >= 0.0f) ? y : 0.01f * y;
    if (residual) y += residual[(size_t)m * N + n];
    Y[(size_t)m * N + n] = y;
    if (Y16) Y16[(size_t)m * N + n] = (_Float16)y;
  }
}

// ---------------------------------------------------------------------------
// GEMM: Y[M,N] = act(X[M,K] @ W[N,K]^T + bias) (+ residual), optional f16 copy.
// Block = 128 threads = 4 waves. Each wave computes a 32x32 tile as 2x2 WMMA
// subtiles (4 wmma per k-step, each A/B fragment reused twice). Block tile:
// 32(M) x 128(N). Requires M%32==0, N%128==0, K%32==0 (true at every call).
// ---------------------------------------------------------------------------
__global__ void gemm_wmma_kernel(const _Float16* __restrict__ X,
                                 const _Float16* __restrict__ W,
                                 const float* __restrict__ bias,
                                 const float* __restrict__ residual,
                                 float* __restrict__ Y,
                                 _Float16* __restrict__ Y16,
                                 int M, int N, int K, int act) {
  const int lane = threadIdx.x & 31;
  const int wave = threadIdx.x >> 5;
  const int n0 = blockIdx.x * 128 + wave * 32;
  const int m0 = blockIdx.y * 32;
  if (n0 >= N || m0 >= M) return;

  v8f acc00 = {}, acc01 = {}, acc10 = {}, acc11 = {};
  for (int k0 = 0; k0 < K; k0 += 32) {
    const v16h a0 = load_A16(X, K, m0, k0, lane);
    const v16h a1 = load_A16(X, K, m0 + 16, k0, lane);
    const v16h b0 = load_B16(W, K, n0, k0, lane);
    const v16h b1 = load_B16(W, K, n0 + 16, k0, lane);
    acc00 = __builtin_amdgcn_wmma_f32_16x16x32_f16(false, a0, false, b0, (short)0, acc00, false, false);
    acc01 = __builtin_amdgcn_wmma_f32_16x16x32_f16(false, a0, false, b1, (short)0, acc01, false, false);
    acc10 = __builtin_amdgcn_wmma_f32_16x16x32_f16(false, a1, false, b0, (short)0, acc10, false, false);
    acc11 = __builtin_amdgcn_wmma_f32_16x16x32_f16(false, a1, false, b1, (short)0, acc11, false, false);
  }
  store_tile(acc00, Y, Y16, bias, residual, N, m0,      n0,      lane, act);
  store_tile(acc01, Y, Y16, bias, residual, N, m0,      n0 + 16, lane, act);
  store_tile(acc10, Y, Y16, bias, residual, N, m0 + 16, n0,      lane, act);
  store_tile(acc11, Y, Y16, bias, residual, N, m0 + 16, n0 + 16, lane, act);
}

// ---------------------------------------------------------------------------
// Small helper kernels
// ---------------------------------------------------------------------------
__global__ void cvt_f32_f16_kernel(const float* __restrict__ s,
                                   _Float16* __restrict__ d, int n) {
  int i = blockIdx.x * 256 + threadIdx.x;
  if (i < n) d[i] = (_Float16)s[i];
}

__global__ void add_vec_kernel(const float* __restrict__ a,
                               const float* __restrict__ b,
                               float* __restrict__ o, int n) {
  int i = blockIdx.x * 256 + threadIdx.x;
  if (i < n) o[i] = a[i] + b[i];
}

// Gather spans (f16 out): out[r][h] = seq[b][ids[r]][h]
__global__ void gather_spans_kernel(const float* __restrict__ seq,
                                    const int* __restrict__ ids,
                                    _Float16* __restrict__ out,
                                    int total, int rows_per_b) {
  int i = blockIdx.x * 256 + threadIdx.x;
  if (i >= total) return;
  const int h = i & (HH - 1);
  const int r = i >> 10;               // H == 1024
  const int b = r / rows_per_b;
  const int s = ids[r];
  out[i] = (_Float16)seq[((size_t)b * SS + s) * HH + h];
}

// LSTM cell: gates = Gin[span*L+t] + Gh[span] (f32); update c (f32), h -> f16
__global__ void lstm_cell_kernel(const float* __restrict__ Gin,
                                 const float* __restrict__ Gh,
                                 _Float16* __restrict__ H,
                                 float* __restrict__ Cst,
                                 _Float16* __restrict__ Yout,
                                 int NS, int L, int t, int dirOff) {
  int i = blockIdx.x * 256 + threadIdx.x;
  if (i >= NS * HCC) return;
  const int span = i >> 9;             // HC == 512
  const int u = i & (HCC - 1);
  const float* gi = Gin + ((size_t)span * L + t) * 2048;
  const float* gh = Gh + (size_t)span * 2048;
  const float gI = gi[u] + gh[u];
  const float gF = gi[512 + u] + gh[512 + u];
  const float gG = gi[1024 + u] + gh[1024 + u];
  const float gO = gi[1536 + u] + gh[1536 + u];
  const float sI = 1.0f / (1.0f + __expf(-gI));
  const float sF = 1.0f / (1.0f + __expf(-gF));
  const float sO = 1.0f / (1.0f + __expf(-gO));
  float c = Cst[i];
  c = sF * c + sI * tanhf(gG);
  const float h = sO * tanhf(c);
  Cst[i] = c;
  H[i] = (_Float16)h;
  Yout[((size_t)span * L + t) * HH + dirOff + u] = (_Float16)h;
}

// pool[span] = concat(hf, hb) = [Y2[span][L-1][0:512], Y2[span][0][512:1024]]
__global__ void extract_pool_kernel(const _Float16* __restrict__ Y2,
                                    _Float16* __restrict__ pool, int NS, int L) {
  int i = blockIdx.x * 256 + threadIdx.x;
  if (i >= NS * HH) return;
  const int span = i >> 10;
  const int u = i & (HH - 1);
  const int t = (u < HCC) ? (L - 1) : 0;
  pool[i] = Y2[((size_t)span * L + t) * HH + u];
}

// Per-(b,t,head) attention over C=8 keys, DH=128 (f32 in, f16 out)
__global__ void attention_kernel(const float* __restrict__ q,
                                 const float* __restrict__ k,
                                 const float* __restrict__ v,
                                 _Float16* __restrict__ o) {
  int i = blockIdx.x * 256 + threadIdx.x;
  if (i >= BB * TT * NHH) return;
  const int head = i & (NHH - 1);
  const int row = i >> 3;              // b*T + t
  const float* qp = q + (size_t)row * HH + head * DHH;
  float sc[CCC];
  float mx = -1e30f;
#pragma unroll
  for (int c = 0; c < CCC; ++c) {
    const float* kp = k + ((size_t)row * CCC + c) * HH + head * DHH;
    float s = 0.0f;
    for (int d = 0; d < DHH; ++d) s += qp[d] * kp[d];
    s *= 0.08838834764831845f;         // 1/sqrt(128)
    sc[c] = s;
    mx = fmaxf(mx, s);
  }
  float den = 0.0f;
#pragma unroll
  for (int c = 0; c < CCC; ++c) { sc[c] = __expf(sc[c] - mx); den += sc[c]; }
  const float inv = 1.0f / den;
  _Float16* op = o + (size_t)row * HH + head * DHH;
  for (int d = 0; d < DHH; ++d) {
    float acc = 0.0f;
#pragma unroll
    for (int c = 0; c < CCC; ++c)
      acc += sc[c] * v[((size_t)row * CCC + c) * HH + head * DHH + d];
    op[d] = (_Float16)(acc * inv);
  }
}

// Row-wise L2 normalize in place (rows of 1024) + f16 copy
__global__ void l2norm_kernel(float* __restrict__ x, _Float16* __restrict__ x16) {
  __shared__ float red[256];
  const int row = blockIdx.x;
  float s = 0.0f;
  for (int j = threadIdx.x; j < HH; j += 256) {
    const float t = x[(size_t)row * HH + j];
    s += t * t;
  }
  red[threadIdx.x] = s;
  __syncthreads();
  for (int off = 128; off > 0; off >>= 1) {
    if (threadIdx.x < off) red[threadIdx.x] += red[threadIdx.x + off];
    __syncthreads();
  }
  const float inv = 1.0f / fmaxf(sqrtf(red[0]), 1e-12f);
  for (int j = threadIdx.x; j < HH; j += 256) {
    const float y = x[(size_t)row * HH + j] * inv;
    x[(size_t)row * HH + j] = y;
    x16[(size_t)row * HH + j] = (_Float16)y;
  }
}

// Final head: out[m][n] = h1[m] . w2[n] + b2[n], N=2, K=256
__global__ void head2_kernel(const float* __restrict__ h1,
                             const float* __restrict__ w2,
                             const float* __restrict__ b2,
                             float* __restrict__ out, int M) {
  int i = blockIdx.x * 256 + threadIdx.x;
  if (i >= M * 2) return;
  const int m = i >> 1, n = i & 1;
  const float* x = h1 + (size_t)m * 256;
  const float* w = w2 + (size_t)n * 256;
  float s = b2[n];
  for (int k2 = 0; k2 < 256; ++k2) s += x[k2] * w[k2];
  out[i] = s;
}

// ---------------------------------------------------------------------------
// Host orchestration
// ---------------------------------------------------------------------------
extern "C" void kernel_launch(void* const* d_in, const int* in_sizes, int n_in,
                              void* d_out, int out_size, void* d_ws, size_t ws_size,
                              hipStream_t stream) {
  (void)in_sizes; (void)n_in; (void)out_size; (void)ws_size;

  const float* seq      = (const float*)d_in[0];
  const int*   t_ids    = (const int*)d_in[1];
  const int*   c_ids    = (const int*)d_in[2];
  const float* t_Wih    = (const float*)d_in[3];
  const float* t_Whh    = (const float*)d_in[4];
  const float* t_bih    = (const float*)d_in[5];
  const float* t_bhh    = (const float*)d_in[6];
  const float* c_Wih    = (const float*)d_in[7];
  const float* c_Whh    = (const float*)d_in[8];
  const float* c_bih    = (const float*)d_in[9];
  const float* c_bhh    = (const float*)d_in[10];
  const float* t_pool_w = (const float*)d_in[11];
  const float* t_pool_b = (const float*)d_in[12];
  const float* c_pool_w = (const float*)d_in[13];
  const float* c_pool_b = (const float*)d_in[14];
  const float* attn_in_w  = (const float*)d_in[15];
  const float* attn_in_b  = (const float*)d_in[16];
  const float* attn_out_w = (const float*)d_in[17];
  const float* attn_out_b = (const float*)d_in[18];
  const float* t_cls1_w = (const float*)d_in[19];
  const float* t_cls1_b = (const float*)d_in[20];
  const float* t_cls2_w = (const float*)d_in[21];
  const float* t_cls2_b = (const float*)d_in[22];
  const float* c_cls1_w = (const float*)d_in[23];
  const float* c_cls1_b = (const float*)d_in[24];
  const float* c_cls2_w = (const float*)d_in[25];
  const float* c_cls2_b = (const float*)d_in[26];
  float* out = (float*)d_out;

  // ---- bump allocator over workspace ----
  char* base = (char*)d_ws;
  size_t off = 0;
  auto alloc = [&](size_t nbytes) -> void* {
    void* p = base + off;
    off += (nbytes + 255) & ~(size_t)255;
    return p;
  };
  auto allocF = [&](size_t n) -> float* { return (float*)alloc(n * sizeof(float)); };
  auto allocH = [&](size_t n) -> _Float16* { return (_Float16*)alloc(n * sizeof(_Float16)); };

  // f16 weight copies
  _Float16* t_Wih_h = allocH((size_t)2 * 2 * 2048 * 1024);
  _Float16* t_Whh_h = allocH((size_t)2 * 2 * 2048 * 512);
  _Float16* c_Wih_h = allocH((size_t)2 * 2 * 2048 * 1024);
  _Float16* c_Whh_h = allocH((size_t)2 * 2 * 2048 * 512);
  _Float16* t_pool_w_h = allocH((size_t)1024 * 1024);
  _Float16* c_pool_w_h = allocH((size_t)1024 * 1024);
  _Float16* attn_in_w_h = allocH((size_t)3072 * 1024);
  _Float16* attn_out_w_h = allocH((size_t)1024 * 1024);
  _Float16* t_cls1_w_h = allocH((size_t)256 * 1024);
  _Float16* c_cls1_w_h = allocH((size_t)256 * 1024);

  const int tNS = BB * TT;             // 128
  const int cNS = BB * TT * CCC;       // 1024

  // f16 activation buffers
  _Float16* t_span = allocH((size_t)tNS * LTT * HH);   // gathered + layer2 out
  _Float16* t_bufA = allocH((size_t)tNS * LTT * HH);   // layer1 out
  _Float16* c_span = allocH((size_t)cNS * LCC * HH);
  _Float16* c_bufA = allocH((size_t)cNS * LCC * HH);
  _Float16* H16 = allocH((size_t)cNS * HCC);
  _Float16* pool_t = allocH((size_t)tNS * HH);
  _Float16* pool_c = allocH((size_t)cNS * HH);
  _Float16* t_emb16 = allocH((size_t)tNS * HH);
  _Float16* c_emb16 = allocH((size_t)cNS * HH);
  _Float16* attn_o16 = allocH((size_t)tNS * HH);
  _Float16* t_res16 = allocH((size_t)tNS * HH);

  // f32 scratch
  float* t_cbias = allocF(2 * 2 * 2048);
  float* c_cbias = allocF(2 * 2 * 2048);
  float* Gin = allocF((size_t)cNS * LCC * 2048);    // shared: covers both phases
  float* Gh  = allocF((size_t)cNS * 2048);
  float* Cst = allocF((size_t)cNS * HCC);
  float* t_emb = allocF((size_t)tNS * HH);          // f32 (residual source)
  float* c_emb = allocF((size_t)cNS * HH);          // f32 (unused downstream; paired out)
  float* qb = allocF((size_t)tNS * HH);
  float* kb = allocF((size_t)cNS * HH);
  float* vb = allocF((size_t)cNS * HH);
  float* t_res = allocF((size_t)tNS * HH);
  float* t_h1 = allocF((size_t)tNS * 256);
  float* c_h1 = allocF((size_t)cNS * 256);

  auto cvt = [&](const float* s, _Float16* d, size_t n) {
    cvt_f32_f16_kernel<<<dim3((unsigned)((n + 255) / 256)), 256, 0, stream>>>(s, d, (int)n);
  };
  auto gemm = [&](const _Float16* X, const _Float16* W, const float* bias,
                  const float* res, float* Y, _Float16* Y16, int M, int N, int K, int act) {
    dim3 g((unsigned)(N / 128), (unsigned)(M / 32));
    gemm_wmma_kernel<<<g, 128, 0, stream>>>(X, W, bias, res, Y, Y16, M, N, K, act);
  };

  // ---- 1. weight conversions + combined biases ----
  cvt(t_Wih, t_Wih_h, (size_t)2 * 2 * 2048 * 1024);
  cvt(t_Whh, t_Whh_h, (size_t)2 * 2 * 2048 * 512);
  cvt(c_Wih, c_Wih_h, (size_t)2 * 2 * 2048 * 1024);
  cvt(c_Whh, c_Whh_h, (size_t)2 * 2 * 2048 * 512);
  cvt(t_pool_w, t_pool_w_h, (size_t)1024 * 1024);
  cvt(c_pool_w, c_pool_w_h, (size_t)1024 * 1024);
  cvt(attn_in_w, attn_in_w_h, (size_t)3072 * 1024);
  cvt(attn_out_w, attn_out_w_h, (size_t)1024 * 1024);
  cvt(t_cls1_w, t_cls1_w_h, (size_t)256 * 1024);
  cvt(c_cls1_w, c_cls1_w_h, (size_t)256 * 1024);
  add_vec_kernel<<<32, 256, 0, stream>>>(t_bih, t_bhh, t_cbias, 8192);
  add_vec_kernel<<<32, 256, 0, stream>>>(c_bih, c_bhh, c_cbias, 8192);

  // ---- 2. gather spans (to f16) ----
  {
    const int nt = tNS * LTT * HH;
    gather_spans_kernel<<<(nt + 255) / 256, 256, 0, stream>>>(seq, t_ids, t_span, nt, TT * LTT);
    const int nc = cNS * LCC * HH;
    gather_spans_kernel<<<(nc + 255) / 256, 256, 0, stream>>>(seq, c_ids, c_span, nc, TT * CCC * LCC);
  }

  // ---- 3. BiLSTM pooling (shared routine) ----
  auto run_bilstm = [&](_Float16* spanBuf, _Float16* bufA, const _Float16* Wih_h,
                        const _Float16* Whh_h, const float* cbias, _Float16* pool,
                        int NS, int L) {
    for (int l = 0; l < 2; ++l) {
      const _Float16* xin = (l == 0) ? spanBuf : bufA;
      _Float16* yout = (l == 0) ? bufA : spanBuf;   // layer2 overwrites span buffer
      for (int d = 0; d < 2; ++d) {
        const _Float16* Wih = Wih_h + (size_t)(l * 2 + d) * 2048 * 1024;
        const _Float16* Whh = Whh_h + (size_t)(l * 2 + d) * 2048 * 512;
        const float* cb = cbias + (size_t)(l * 2 + d) * 2048;
        // Gin = xin @ Wih^T + (bih+bhh): M = NS*L, K = 1024, N = 2048
        gemm(xin, Wih, cb, nullptr, Gin, nullptr, NS * L, 2048, 1024, 0);
        hipMemsetAsync(H16, 0, (size_t)NS * HCC * sizeof(_Float16), stream);
        hipMemsetAsync(Cst, 0, (size_t)NS * HCC * sizeof(float), stream);
        for (int s = 0; s < L; ++s) {
          // Gh = H @ Whh^T: M = NS, K = 512, N = 2048
          gemm(H16, Whh, nullptr, nullptr, Gh, nullptr, NS, 2048, 512, 0);
          const int t = d ? (L - 1 - s) : s;
          const int nthr = NS * HCC;
          lstm_cell_kernel<<<(nthr + 255) / 256, 256, 0, stream>>>(
              Gin, Gh, H16, Cst, yout, NS, L, t, d * HCC);
        }
      }
    }
    const int nthr = NS * HH;
    extract_pool_kernel<<<(nthr + 255) / 256, 256, 0, stream>>>(spanBuf, pool, NS, L);
  };

  run_bilstm(t_span, t_bufA, t_Wih_h, t_Whh_h, t_cbias, pool_t, tNS, LTT);
  run_bilstm(c_span, c_bufA, c_Wih_h, c_Whh_h, c_cbias, pool_c, cNS, LCC);

  // ---- 4. pool projections (leaky relu), keep f32 + f16 ----
  gemm(pool_t, t_pool_w_h, t_pool_b, nullptr, t_emb, t_emb16, tNS, HH, HH, 1);
  gemm(pool_c, c_pool_w_h, c_pool_b, nullptr, c_emb, c_emb16, cNS, HH, HH, 1);

  // ---- 5. attention projections ----
  gemm(t_emb16, attn_in_w_h, attn_in_b, nullptr, qb, nullptr, tNS, HH, HH, 0);
  gemm(c_emb16, attn_in_w_h + (size_t)1024 * 1024, attn_in_b + 1024, nullptr, kb, nullptr, cNS, HH, HH, 0);
  gemm(c_emb16, attn_in_w_h + (size_t)2048 * 1024, attn_in_b + 2048, nullptr, vb, nullptr, cNS, HH, HH, 0);

  attention_kernel<<<(BB * TT * NHH + 255) / 256, 256, 0, stream>>>(qb, kb, vb, attn_o16);

  // ---- 6. out projection + residual, L2 norm ----
  gemm(attn_o16, attn_out_w_h, attn_out_b, t_emb, t_res, nullptr, tNS, HH, HH, 0);
  l2norm_kernel<<<tNS, 256, 0, stream>>>(t_res, t_res16);

  // ---- 7. classifier heads ----
  gemm(t_res16, t_cls1_w_h, t_cls1_b, nullptr, t_h1, nullptr, tNS, 256, HH, 1);
  head2_kernel<<<(tNS * 2 + 255) / 256, 256, 0, stream>>>(t_h1, t_cls2_w, t_cls2_b, out, tNS);

  gemm(c_emb16, c_cls1_w_h, c_cls1_b, nullptr, c_h1, nullptr, cNS, 256, HH, 1);
  head2_kernel<<<(cNS * 2 + 255) / 256, 256, 0, stream>>>(c_h1, c_cls2_w, c_cls2_b,
                                                          out + BB * TT * 2, cNS);
}
